// Net2_56736517980179
// MI455X (gfx1250) — matrix operand order repaired and checked by reading, not compiled, not run on hardware
//
#include <hip/hip_runtime.h>

// ---------------------------------------------------------------------------
// Problem constants (reference): B=8192, D_IN=512, O=16, C=8, H=32, F=576
// All heavy GEMM operands are f16; WMMA = v_wmma_f32_16x16x32_f16.
// Tile staging: GLOBAL_LOAD_ASYNC_TO_LDS_B128, double-buffered (ASYNCcnt).
// GEMM epilogue drains via GLOBAL_STORE_ASYNC_FROM_LDS_B128.
// ---------------------------------------------------------------------------

typedef __attribute__((ext_vector_type(4)))  float    v4f;
typedef __attribute__((ext_vector_type(8)))  float    v8f;
typedef __attribute__((ext_vector_type(2)))  __fp16   v2fp;   // cvt_pkrtz result type
typedef __attribute__((ext_vector_type(4)))  _Float16 v4h;
typedef __attribute__((ext_vector_type(16))) _Float16 v16h;

union FragU { v4f f4[2]; v16h h; };
union PkU  { v2fp p[2]; v4h w; };

// A-matrix fragment (16x32 f16): lane kh=lane>>4 selects K groups
// {kh*8..+7} and {16+kh*8..+7}; row = lane&15 (ISA 7.12.2).
__device__ __forceinline__ v16h fragA(const _Float16* row, int kh) {
  FragU u;
  u.f4[0] = *(const v4f*)(row + kh * 8);
  u.f4[1] = *(const v4f*)(row + 16 + kh * 8);
  return u.h;
}
// B-matrix fragment (32x16 f16) staged [n][k]: lane kh=lane>>4 selects K groups
// {kh*8..+7} and {16+kh*8..+7} (same order as fragA).
__device__ __forceinline__ v16h fragB(const _Float16* row, int kh) {
  FragU u;
  u.f4[0] = *(const v4f*)(row + kh * 8);
  u.f4[1] = *(const v4f*)(row + 16 + kh * 8);
  return u.h;
}

#define WMMA_F16(A, B, C) \
  __builtin_amdgcn_wmma_f32_16x16x32_f16(false, (A), false, (B), (short)0, (C), false, false)

// Async DMA global -> LDS, 16B per lane, tracked by ASYNCcnt (ISA 15.18.3).
__device__ __forceinline__ void async_copy16(unsigned ldsOff, const void* gaddr) {
  asm volatile("global_load_async_to_lds_b128 %0, %1, off"
               :: "v"(ldsOff), "v"((unsigned long long)gaddr) : "memory");
}
// Async DMA LDS -> global, 16B per lane (addr, lds-src, off).
__device__ __forceinline__ void async_store16(const void* gaddr, unsigned ldsOff) {
  asm volatile("global_store_async_from_lds_b128 %0, %1, off"
               :: "v"((unsigned long long)gaddr), "v"(ldsOff) : "memory");
}
__device__ __forceinline__ void wait_async0() {
#if __has_builtin(__builtin_amdgcn_s_wait_asynccnt)
  __builtin_amdgcn_s_wait_asynccnt(0);
#else
  asm volatile("s_wait_asynccnt 0x0" ::: "memory");
#endif
}
__device__ __forceinline__ unsigned lds_off(const void* p) {
  return (unsigned)(unsigned long long)p;
}

// ---------------------------------------------------------------------------
// x (f32) -> xh (f16), packed v_cvt_pk_rtz, 4 elements/thread.
// ---------------------------------------------------------------------------
__global__ __launch_bounds__(256) void cvt_f32_f16_kernel(
    const float* __restrict__ in, _Float16* __restrict__ out)
{
  long i = ((long)blockIdx.x * 256 + threadIdx.x) * 4;
  v4f v = *(const v4f*)(in + i);
  PkU u;
  u.p[0] = __builtin_amdgcn_cvt_pkrtz(v.x, v.y);
  u.p[1] = __builtin_amdgcn_cvt_pkrtz(v.z, v.w);
  *(v4h*)(out + i) = u.w;
}

// ---------------------------------------------------------------------------
// Transpose-convert weights: out(N,K) f16 = in(K,N) f32, 32x32 LDS tiles.
// All shapes used are multiples of 32 -> no guards. blockIdx.z batches.
// ---------------------------------------------------------------------------
__global__ __launch_bounds__(256) void transpose_cvt_kernel(
    const float* __restrict__ in, int N, _Float16* __restrict__ out, int K,
    long inStrideZ, long outStrideZ)
{
  __shared__ float tile[32][33];
  const float* inz = in + (long)blockIdx.z * inStrideZ;
  _Float16* outz = out + (long)blockIdx.z * outStrideZ;
  int k0 = blockIdx.x * 32, n0 = blockIdx.y * 32;
  int tx = threadIdx.x & 31, ty = threadIdx.x >> 5;  // 32 x 8
  #pragma unroll
  for (int r = ty; r < 32; r += 8)
    tile[r][tx] = inz[(long)(k0 + r) * N + n0 + tx];
  __syncthreads();
  #pragma unroll
  for (int r = ty; r < 32; r += 8)
    outz[(long)(n0 + r) * K + k0 + tx] = (_Float16)tile[tx][r];
}

// ---------------------------------------------------------------------------
// Tiled GEMM, all-f16: C = act(A @ BT^T + bias)
//   A (M,K) f16 row-major, BT (N,K) f16 row-major, C (M,N) f16.
//   Tile 128x64, BK=32, 8 waves. Double-buffered async-to-LDS staging;
//   LDS-staged epilogue drained by async store-from-LDS.
// ---------------------------------------------------------------------------
__global__ __launch_bounds__(256) void gemm_h16_wmma_kernel(
    const _Float16* __restrict__ A, int lda,
    const _Float16* __restrict__ BT, int ldb,
    const float* __restrict__ bias,
    _Float16* __restrict__ C, int ldc,
    int K,
    long btStrideZ, long biasStrideZ, long cOffZ,
    int relu)
{
  // pool: As0 8K | As1 8K | Bs0 4K | Bs1 4K ; C tile (16K) reuses As0+As1
  __shared__ __align__(16) char smem[24576];
  _Float16* As0 = (_Float16*)smem;
  _Float16* As1 = (_Float16*)(smem + 8192);
  _Float16* Bs0 = (_Float16*)(smem + 16384);
  _Float16* Bs1 = (_Float16*)(smem + 20480);
  _Float16* Cs  = (_Float16*)smem;

  const _Float16* BTz = BT + (long)blockIdx.z * btStrideZ;
  const float* bz = bias + (long)blockIdx.z * biasStrideZ;
  _Float16* Cz = C + (long)blockIdx.z * cOffZ;

  const int rowBase = blockIdx.x * 128;
  const int colBase = blockIdx.y * 64;
  const int tid  = threadIdx.x;
  const int wave = tid >> 5, lane = tid & 31;
  const int wm = wave >> 1, wn = wave & 1;   // 4x2 waves -> 32x32 each
  const int kh = lane >> 4, ml = lane & 15;

  const unsigned asB0 = lds_off(As0), asB1 = lds_off(As1);
  const unsigned bsB0 = lds_off(Bs0), bsB1 = lds_off(Bs1);

  const char* Arow = (const char*)(A + (long)rowBase * lda);
  const char* Brow = (const char*)(BTz + (long)colBase * ldb);

  // stage tile ki: A 512 x 16B chunks (2/thread), B 256 x 16B chunks (1/thread)
  const int ra = tid >> 2, ca = (tid & 3) * 16;   // A chunk 0
  const int ra2 = (tid + 256) >> 2;               // A chunk 1 (same ca)

  auto stage = [&](int ki, unsigned asBuf, unsigned bsBuf) {
    long kb = (long)ki * 64;  // 32 halves = 64 bytes
    async_copy16(asBuf + ra * 64 + ca,  Arow + (long)ra  * lda * 2 + kb + ca);
    async_copy16(asBuf + ra2 * 64 + ca, Arow + (long)ra2 * lda * 2 + kb + ca);
    async_copy16(bsBuf + ra * 64 + ca,  Brow + (long)ra  * ldb * 2 + kb + ca);
  };

  v8f acc00 = {}, acc01 = {}, acc10 = {}, acc11 = {};

  const int nk = K >> 5;
  stage(0, asB0, bsB0);
  for (int ki = 0; ki < nk; ++ki) {
    const int cur = ki & 1;
    wait_async0();        // own copies for tile ki complete
    __syncthreads();      // everyone's copies visible; prev buffer free
    if (ki + 1 < nk) stage(ki + 1, cur ? asB0 : asB1, cur ? bsB0 : bsB1);

    const _Float16* Ac = cur ? As1 : As0;
    const _Float16* Bc = cur ? Bs1 : Bs0;
    v16h a0 = fragA(&Ac[(wm * 32 + ml) * 32], kh);
    v16h a1 = fragA(&Ac[(wm * 32 + 16 + ml) * 32], kh);
    v16h b0 = fragB(&Bc[(wn * 32 + ml) * 32], kh);
    v16h b1 = fragB(&Bc[(wn * 32 + 16 + ml) * 32], kh);

    acc00 = WMMA_F16(a0, b0, acc00);
    acc01 = WMMA_F16(a0, b1, acc01);
    acc10 = WMMA_F16(a1, b0, acc10);
    acc11 = WMMA_F16(a1, b1, acc11);
  }
  __syncthreads();  // all waves done with A/B buffers -> reuse as C tile

  // Epilogue into LDS: lane col = ml, rows = r + 8*kh (ISA 7.12.2)
  #pragma unroll
  for (int s = 0; s < 2; ++s) {
    #pragma unroll
    for (int t = 0; t < 2; ++t) {
      v8f acc = (s == 0) ? (t == 0 ? acc00 : acc01)
                         : (t == 0 ? acc10 : acc11);
      int col = wn * 32 + t * 16 + ml;
      float bv = bz[colBase + col];
      #pragma unroll
      for (int r = 0; r < 8; ++r) {
        int row = wm * 32 + s * 16 + (r + 8 * kh);
        float v = acc[r] + bv;
        if (relu) v = fmaxf(v, 0.f);
        Cs[row * 64 + col] = (_Float16)v;
      }
    }
  }
  __syncthreads();

  // Drain C tile (128 rows x 128B) with async store-from-LDS, 4 chunks/thread
  const char* Cbase = (const char*)(Cz + (long)rowBase * ldc + colBase);
  const unsigned csBase = lds_off(Cs);
  #pragma unroll
  for (int q = 0; q < 4; ++q) {
    int u = tid + q * 256;
    int r = u >> 3, ch = (u & 7) * 16;
    async_store16(Cbase + (long)r * ldc * 2 + ch, csBase + r * 128 + ch);
  }
  wait_async0();
}

// ---------------------------------------------------------------------------
// Batch-independent attention constants: att (16x512), attT (16x512). 1 block.
// ---------------------------------------------------------------------------
__global__ __launch_bounds__(256) void consts_kernel(
    const float* __restrict__ coef_m, const float* __restrict__ W_c,
    const float* __restrict__ b_c,    const float* __restrict__ W_ca,
    const float* __restrict__ W_oa,   const float* __restrict__ W_t,
    float* __restrict__ att, float* __restrict__ attT)
{
  __shared__ float sS[16 * 16 * 32];
  __shared__ float sCS[16 * 32];
  const int tid = threadIdx.x;
  const int x = tid >> 4, y = tid & 15;

  float denom = 0.f;
  for (int c = 0; c < 8; ++c) {
    float cm = coef_m[c * 256 + x * 16 + y];
    #pragma unroll
    for (int h = 0; h < 32; ++h) {
      float pre = cm * W_c[c * 32 + h] + b_c[c * 32 + h];
      float f = pre > 0.f ? pre : (__expf(pre) - 1.f);
      denom += __expf(f);
    }
  }
  float sacc[32];
  #pragma unroll
  for (int k = 0; k < 32; ++k) sacc[k] = 0.f;
  for (int c = 0; c < 8; ++c) {
    float cm = coef_m[c * 256 + x * 16 + y];
    float fv[32];
    #pragma unroll
    for (int h = 0; h < 32; ++h) {
      float pre = cm * W_c[c * 32 + h] + b_c[c * 32 + h];
      fv[h] = pre > 0.f ? pre : (__expf(pre) - 1.f);
    }
    #pragma unroll
    for (int k = 0; k < 32; ++k) {
      float g = 0.f;
      for (int h = 0; h < 32; ++h) g += fv[h] * W_ca[c * 1024 + h * 32 + k];
      sacc[k] += __expf(fv[k]) * g;
    }
  }
  float inv = 1.f / (8.f * denom);
  #pragma unroll
  for (int k = 0; k < 32; ++k) sS[tid * 32 + k] = sacc[k] * inv;
  __syncthreads();

  for (int u = tid; u < 16 * 32; u += 256) {
    int xx = u >> 5, k = u & 31;
    float s = 0.f;
    for (int yy = 0; yy < 16; ++yy) s += __expf(sS[(xx * 16 + yy) * 32 + k]);
    sCS[u] = s;
  }
  __syncthreads();

  for (int u = tid; u < 16 * 512; u += 256) {
    int o = u >> 9, rest = u & 511;
    int i = rest >> 5, k = rest & 31;
    float dot = 0.f;
    for (int h = 0; h < 32; ++h) dot += sS[(o * 16 + i) * 32 + h] * W_oa[i * 1024 + h * 32 + k];
    att[u] = __expf(sS[(o * 16 + i) * 32 + k]) / sCS[o * 32 + k] * dot;
  }
  __threadfence();
  __syncthreads();

  for (int u = tid; u < 16 * 512; u += 256) {
    int n = u >> 9, f = u & 511;
    float a = 0.f;
    for (int o = 0; o < 16; ++o) a += W_t[o * 16 + n] * att[o * 512 + f];
    attT[u] = a;
  }
}

// Kc(n,j) = sum_f attT(n,f) * W_g[64+f, j]  -> (16 x 576) f32
__global__ void kconst_kernel(const float* __restrict__ attT,
                              const float* __restrict__ Wg,
                              float* __restrict__ Kc)
{
  int j = blockIdx.x * 256 + threadIdx.x;
  if (j >= 16 * 576) return;
  int n = j / 576, col = j % 576;
  float acc = 0.f;
  for (int f = 0; f < 512; ++f) acc += attT[n * 512 + f] * Wg[(long)(64 + f) * 576 + col];
  Kc[j] = acc;
}

// Z(b,n,f) = sum_o hid(b,o,f) * W_t[o,n] ; f16 in/out, one block per batch b.
__global__ __launch_bounds__(256) void z_kernel(const _Float16* __restrict__ hid,
                                                const float* __restrict__ W_t,
                                                _Float16* __restrict__ Z)
{
  __shared__ float hs[1024];
  __shared__ float wt[256];
  long b = blockIdx.x;
  int tid = threadIdx.x;
  for (int i = tid; i < 1024; i += 256) hs[i] = (float)hid[b * 1024 + i];
  wt[tid] = W_t[tid];
  __syncthreads();
  for (int u = tid; u < 1024; u += 256) {
    int n = u >> 6, f = u & 63;
    float a = 0.f;
    #pragma unroll
    for (int o = 0; o < 16; ++o) a += hs[o * 64 + f] * wt[o * 16 + n];
    Z[b * 1024 + u] = (_Float16)a;
  }
}

// ---------------------------------------------------------------------------
// Fused final: out(b,n) = prelu(b_m + sum_j sigmoid(Z@Wg[:64,j] + Kc(n,j))*Wm[j])
// M=131072 rows, K=64, N=576 (9 stages of 64, double-buffered Ws).
// ---------------------------------------------------------------------------
__global__ __launch_bounds__(256) void final_kernel(
    const _Float16* __restrict__ Z, const _Float16* __restrict__ WgT,
    const float* __restrict__ Kc, const float* __restrict__ Wm,
    const float* __restrict__ bm, const float* __restrict__ alpha,
    float* __restrict__ out)
{
  // Zs 16K | Ws0 8K | Ws1 8K
  __shared__ __align__(16) char smemF[32768];
  _Float16* Zs  = (_Float16*)smemF;
  _Float16* Ws0 = (_Float16*)(smemF + 16384);
  _Float16* Ws1 = (_Float16*)(smemF + 24576);

  const int rowBase = blockIdx.x * 128;
  const int tid  = threadIdx.x;
  const int wave = tid >> 5, lane = tid & 31;
  const int kh = lane >> 4, ml = lane & 15;

  const unsigned zsBase = lds_off(Zs);
  const unsigned wsB0 = lds_off(Ws0), wsB1 = lds_off(Ws1);

  const char* Wrow = (const char*)WgT;  // (576,64) f16, 128B rows
  const int rw = tid >> 1, cw = (tid & 1) * 16;  // 128 rows x 8 chunks = 1024 (4/thread)
  auto stageW = [&](int ns, unsigned wsBuf) {
    long base = (long)ns * 64 * 128;
    async_copy16(wsBuf + rw * 128 + cw,      Wrow + base + (long)rw * 128 + cw);
    async_copy16(wsBuf + rw * 128 + cw + 32, Wrow + base + (long)rw * 128 + cw + 32);
    async_copy16(wsBuf + rw * 128 + cw + 64, Wrow + base + (long)rw * 128 + cw + 64);
    async_copy16(wsBuf + rw * 128 + cw + 96, Wrow + base + (long)rw * 128 + cw + 96);
  };

  {  // prologue: stage Z tile (128 rows x 128B = 1024 chunks, 4/thread) + Ws stage 0
    const char* Zbase = (const char*)(Z + (long)rowBase * 64);
    #pragma unroll
    for (int q = 0; q < 4; ++q) {
      int u = tid + q * 256;
      int r = u >> 3, ch = (u & 7) * 16;
      async_copy16(zsBase + r * 128 + ch, Zbase + (long)r * 128 + ch);
    }
    stageW(0, wsB0);
  }

  float partial[8];
  #pragma unroll
  for (int r = 0; r < 8; ++r) partial[r] = 0.f;

  v16h a0, a1;
  for (int ns = 0; ns < 9; ++ns) {
    const int cur = ns & 1;
    wait_async0();
    __syncthreads();
    if (ns + 1 < 9) stageW(ns + 1, cur ? wsB0 : wsB1);
    if (ns == 0) {
      a0 = fragA(&Zs[(wave * 16 + ml) * 64 + 0],  kh);
      a1 = fragA(&Zs[(wave * 16 + ml) * 64 + 32], kh);
    }
    const _Float16* Wc = cur ? Ws1 : Ws0;

    #pragma unroll
    for (int t = 0; t < 4; ++t) {
      v8f acc = {};
      v16h b0 = fragB(&Wc[(t * 16 + ml) * 64 + 0],  kh);
      v16h b1 = fragB(&Wc[(t * 16 + ml) * 64 + 32], kh);
      acc = WMMA_F16(a0, b0, acc);
      acc = WMMA_F16(a1, b1, acc);

      int j = ns * 64 + t * 16 + ml;
      float wmj = Wm[j];
      #pragma unroll
      for (int r = 0; r < 8; ++r) {
        int m = r + 8 * kh;  // row within 16-tile == node index n
        float v = acc[r] + Kc[m * 576 + j];
        float sig = 1.f / (1.f + __expf(-v));
        partial[r] += sig * wmj;
      }
    }
  }

  #pragma unroll
  for (int r = 0; r < 8; ++r) {
    float p = partial[r];
    p += __shfl_xor(p, 1, 32);
    p += __shfl_xor(p, 2, 32);
    p += __shfl_xor(p, 4, 32);
    p += __shfl_xor(p, 8, 32);
    partial[r] = p;
  }
  if (ml == 0) {
    float a = alpha[0], bb = bm[0];
    #pragma unroll
    for (int r = 0; r < 8; ++r) {
      int row = rowBase + wave * 16 + 8 * kh + r;
      float v = partial[r] + bb;
      out[row] = v >= 0.f ? v : a * v;
    }
  }
}

// ---------------------------------------------------------------------------
// Workspace layout (bytes), all 16B aligned, total ~45.8 MB:
//   xh   @ 0          (8 MiB)      h1h @ 8388608  (16 MiB)  | Zh reuses h1h
//   h2h  @ 25165824   (2 MiB)      hidh@ 27262976 (16 MiB)
//   WinT @ 44040192 (1M)  W1T @ 45088768 (256K)  WoutT @ 45350912 (256K)
//   WgT  @ 45613056 (72K) att @ 45686784  attT @ 45719552  Kc @ 45752320
// ---------------------------------------------------------------------------
extern "C" void kernel_launch(void* const* d_in, const int* in_sizes, int n_in,
                              void* d_out, int out_size, void* d_ws, size_t ws_size,
                              hipStream_t stream) {
  const float* x      = (const float*)d_in[0];
  const float* coef_m = (const float*)d_in[1];
  const float* W_in   = (const float*)d_in[2];
  const float* b_in   = (const float*)d_in[3];
  const float* W1     = (const float*)d_in[4];
  const float* b1     = (const float*)d_in[5];
  const float* W_out  = (const float*)d_in[6];
  const float* b_out  = (const float*)d_in[7];
  const float* W_c    = (const float*)d_in[8];
  const float* b_c    = (const float*)d_in[9];
  const float* W_ca   = (const float*)d_in[10];
  const float* W_oa   = (const float*)d_in[11];
  const float* W_t    = (const float*)d_in[12];
  const float* W_g    = (const float*)d_in[13];
  const float* W_m    = (const float*)d_in[14];
  const float* b_m    = (const float*)d_in[15];
  const float* alpha  = (const float*)d_in[16];

  char* wsb = (char*)d_ws;
  _Float16* xh    = (_Float16*)(wsb + 0);
  _Float16* h1h   = (_Float16*)(wsb + 8388608L);
  _Float16* h2h   = (_Float16*)(wsb + 25165824L);
  _Float16* hidh  = (_Float16*)(wsb + 27262976L);
  _Float16* Zh    = h1h;  // h1h dead after gemm2
  _Float16* WinT  = (_Float16*)(wsb + 44040192L);
  _Float16* W1T   = (_Float16*)(wsb + 45088768L);
  _Float16* WoutT = (_Float16*)(wsb + 45350912L);
  _Float16* WgT   = (_Float16*)(wsb + 45613056L);
  float* att  = (float*)(wsb + 45686784L);
  float* attT = (float*)(wsb + 45719552L);
  float* Kc   = (float*)(wsb + 45752320L);

  // batch-independent constants
  consts_kernel<<<1, 256, 0, stream>>>(coef_m, W_c, b_c, W_ca, W_oa, W_t, att, attT);
  kconst_kernel<<<36, 256, 0, stream>>>(attT, W_g, Kc);

  // one-time f16 conversions / weight transposes
  cvt_f32_f16_kernel<<<4096, 256, 0, stream>>>(x, xh);                    // 8192x512
  transpose_cvt_kernel<<<dim3(16, 32, 1),  256, 0, stream>>>(W_in, 1024, WinT, 512, 0, 0);
  transpose_cvt_kernel<<<dim3(32, 4, 1),   256, 0, stream>>>(W1, 128, W1T, 1024, 0, 0);
  transpose_cvt_kernel<<<dim3(4, 2, 16),   256, 0, stream>>>(W_out, 64, WoutT, 128,
                                                             128L * 64, 64L * 128);
  transpose_cvt_kernel<<<dim3(2, 18, 1),   256, 0, stream>>>(W_g, 576, WgT, 64, 0, 0);

  // h1 = relu(x @ W_in + b_in)   (8192,512)x(512,1024)
  gemm_h16_wmma_kernel<<<dim3(64, 16, 1), 256, 0, stream>>>(
      xh, 512, WinT, 512, b_in, h1h, 1024, 512, 0, 0, 0, 1);
  // h2 = relu(h1 @ W1 + b1)      (8192,1024)x(1024,128)
  gemm_h16_wmma_kernel<<<dim3(64, 2, 1), 256, 0, stream>>>(
      h1h, 1024, W1T, 1024, b1, h2h, 128, 1024, 0, 0, 0, 1);
  // hid[:,o,:] = relu(h2 @ W_out[o] + b_out[o]), 16 batched (8192,128)x(128,64)
  gemm_h16_wmma_kernel<<<dim3(64, 1, 16), 256, 0, stream>>>(
      h2h, 128, WoutT, 128, b_out, hidh, 1024, 128,
      /*btStrideZ=*/64L * 128, /*biasStrideZ=*/64, /*cOffZ=*/64, 1);

  // Z(b,n,f) = sum_o hid(b,o,f) * W_t[o,n]
  z_kernel<<<8192, 256, 0, stream>>>(hidh, W_t, Zh);

  // fused sigmoid-GEMM + W_m reduction + PReLU
  final_kernel<<<1024, 256, 0, stream>>>(Zh, WgT, Kc, W_m, b_m, alpha, (float*)d_out);
}
